// MoleculeNet_88983132439267
// MI455X (gfx1250) — matrix operand order repaired and checked by reading, not compiled, not run on hardware
//
#include <hip/hip_runtime.h>
#include <hip/hip_bf16.h>

typedef __attribute__((ext_vector_type(16))) _Float16 v16h;
typedef __attribute__((ext_vector_type(8)))  _Float16 v8h;
typedef __attribute__((ext_vector_type(8)))  float    v8f;

#define N_ATOMS 2097152
#define N_MOLS  65536
#define N_ELEM  3
#define D_IN    128
#define H0      64
#define H1      32

#define BLK_THREADS 256
#define WAVES 8
#define TILE_M 16
#define ATOMS_PER_BLOCK 2048
#define MAIN_BLOCKS (N_ATOMS / ATOMS_PER_BLOCK)
#define ITERS (ATOMS_PER_BLOCK / (WAVES * TILE_M))

#define SHUF16(lo, hi) __builtin_shufflevector((lo), (hi), 0,1,2,3,4,5,6,7,8,9,10,11,12,13,14,15)

// ---------------------------------------------------------------- utilities

__global__ void molnet_zero_ws(float* __restrict__ p) {
    int i = blockIdx.x * blockDim.x + threadIdx.x;
    if (i < 2 * N_MOLS) p[i] = 0.0f;
}

// W0 [3][128][64] f32 (row-major k,n)  ->  W0t [3][64][128] f16 ([e][n][k], k contiguous)
__global__ void molnet_prep_w0(const float* __restrict__ W0, _Float16* __restrict__ W0t) {
    int i = blockIdx.x * blockDim.x + threadIdx.x;
    if (i < N_ELEM * D_IN * H0) {
        int e = i / (D_IN * H0);
        int rem = i % (D_IN * H0);
        int k = rem / H0;
        int n = rem % H0;
        W0t[((e * H0) + n) * D_IN + k] = (_Float16)W0[i];
    }
}

__global__ void molnet_finalize(const float* __restrict__ sums,
                                const float* __restrict__ counts,
                                float* __restrict__ out) {
    int m = blockIdx.x * blockDim.x + threadIdx.x;
    if (m < N_MOLS) out[m] = sums[m] / fmaxf(counts[m], 1.0f);
}

__device__ __forceinline__ float softplus_fast(float x) {
    // log1p(exp(x)) == max(x,0) + log(1 + exp(-|x|)); 1+t in (1,2] -> plain log fine
    return fmaxf(x, 0.0f) + __logf(1.0f + __expf(-fabsf(x)));
}

// ---------------------------------------------------------------- main kernel

__global__ __launch_bounds__(BLK_THREADS)
void molnet_main(const float* __restrict__ feats,      // [N_ATOMS][128]
                 const int*   __restrict__ elemIdx,    // [N_ATOMS]
                 const int*   __restrict__ molIdx,     // [N_ATOMS]
                 const _Float16* __restrict__ W0t,     // [3][64][128] f16, k contiguous
                 const float* __restrict__ W1,         // [3][64][32]
                 const float* __restrict__ b0,         // [3][64]
                 const float* __restrict__ b1,         // [3][32]
                 const float* __restrict__ W2,         // [3][32]
                 const float* __restrict__ b2,         // [3]
                 float* __restrict__ sums,
                 float* __restrict__ counts)
{
    __shared__ _Float16 sW1t[N_ELEM][H1][H0];   // [e][n][k], 12 KB
    __shared__ _Float16 sH [WAVES][TILE_M][H0]; // layer-0 activations, 16 KB
    __shared__ float    sH2[WAVES][TILE_M][H1]; // layer-1 activations, 16 KB
    __shared__ float    sB0[N_ELEM * H0];
    __shared__ float    sB1[N_ELEM * H1];
    __shared__ float    sW2[N_ELEM * H1];
    __shared__ float    sB2[N_ELEM];
    __shared__ int      sElem[WAVES][TILE_M];

    const int tid = threadIdx.x;

    // ---- stage small weights into LDS (W1 transposed + converted to f16) ----
    for (int i = tid; i < N_ELEM * H0 * H1; i += BLK_THREADS) {
        int e = i / (H0 * H1);
        int rem = i % (H0 * H1);
        int k = rem / H1;
        int n = rem % H1;
        sW1t[e][n][k] = (_Float16)W1[i];
    }
    for (int i = tid; i < N_ELEM * H0; i += BLK_THREADS) sB0[i] = b0[i];
    if (tid < N_ELEM * H1) { sB1[tid] = b1[tid]; sW2[tid] = W2[tid]; }
    if (tid < N_ELEM)      sB2[tid] = b2[tid];
    __syncthreads();

    const int wave  = tid >> 5;
    const int lane  = tid & 31;
    const int mrow  = lane & 15;          // A-row / B-col / C-col within 16-tile
    const int koff  = (lane >> 4) << 3;   // A frag: 0 or 8  (half-wave k offset)
    const int bk    = (lane >> 4) << 4;   // 0 or 16, unused (B frags use koff)
    const int base8 = (lane >> 4) << 3;   // C frag: rows base (0 or 8)
    const long blockBase = (long)blockIdx.x * ATOMS_PER_BLOCK;

    for (int it = 0; it < ITERS; ++it) {
        const long tileBase = blockBase + (long)it * (WAVES * TILE_M) + wave * TILE_M;

        // Opaque zero, re-defined every iteration: defeats LICM of the
        // loop-invariant B-fragment loads (they must stream from WGP$/LDS,
        // not be hoisted into hundreds of VGPRs).
        int opaque = 0;
        asm volatile("" : "+s"(opaque));
        const _Float16* W0base = W0t + opaque;
        const _Float16* W1base = &sW1t[0][0][0] + opaque;

        if (lane < 16) sElem[wave][lane] = elemIdx[tileBase + lane];

        const float* arow = feats + (tileBase + mrow) * D_IN;
        if (it + 1 < ITERS)
            __builtin_prefetch(arow + WAVES * TILE_M * D_IN, 0, 1); // global_prefetch_b8

        // ---- A fragments: 16x128 f32 tile -> f16, per 16x16x32 WMMA layout ----
        v16h afrag[4];
#pragma unroll
        for (int kq = 0; kq < 4; ++kq) {
            const float* p = arow + kq * 32 + koff;
            float4 f0 = *(const float4*)(p);
            float4 f1 = *(const float4*)(p + 4);
            float4 f2 = *(const float4*)(p + 16);
            float4 f3 = *(const float4*)(p + 20);
            v16h a = { (_Float16)f0.x,(_Float16)f0.y,(_Float16)f0.z,(_Float16)f0.w,
                       (_Float16)f1.x,(_Float16)f1.y,(_Float16)f1.z,(_Float16)f1.w,
                       (_Float16)f2.x,(_Float16)f2.y,(_Float16)f2.z,(_Float16)f2.w,
                       (_Float16)f3.x,(_Float16)f3.y,(_Float16)f3.z,(_Float16)f3.w };
            afrag[kq] = a;
        }

        int er[8];
#pragma unroll
        for (int q = 0; q < 8; ++q) er[q] = sElem[wave][base8 + q];

        // ---- layer 0: nt outer, all 3 element weight-sets for this 16-col
        //      group (24 live acc VGPRs), branchless select, one store ----
#pragma unroll
        for (int nt = 0; nt < 4; ++nt) {
            v8f c[N_ELEM];
#pragma unroll
            for (int e = 0; e < N_ELEM; ++e) {
                v8f acc = {};
#pragma unroll
                for (int kq = 0; kq < 4; ++kq) {
                    const _Float16* bp = W0base + ((e * H0) + nt * 16 + mrow) * D_IN + kq * 32 + koff;
                    v8h blo = *(const v8h*)bp;
                    v8h bhi = *(const v8h*)(bp + 16);
                    v16h b = SHUF16(blo, bhi);
                    acc = __builtin_amdgcn_wmma_f32_16x16x32_f16(
                              false, afrag[kq], false, b, (short)0, acc, false, false);
                }
                c[e] = acc;
            }
            const int n = nt * 16 + mrow;
            const float bb0 = sB0[0 * H0 + n];
            const float bb1 = sB0[1 * H0 + n];
            const float bb2 = sB0[2 * H0 + n];
#pragma unroll
            for (int q = 0; q < 8; ++q) {
                const int e = er[q];
                float x = (e == 0 ? c[0][q] : (e == 1 ? c[1][q] : c[2][q]))
                        + (e == 0 ? bb0 : (e == 1 ? bb1 : bb2));
                sH[wave][base8 + q][n] = (_Float16)softplus_fast(x);
            }
        }

        // ---- layer 1 A fragments from LDS (transpose via round-trip) ----
        v16h a1[2];
#pragma unroll
        for (int kq = 0; kq < 2; ++kq) {
            const _Float16* hp = &sH[wave][mrow][kq * 32 + koff];
            v8h lo = *(const v8h*)hp;
            v8h hi = *(const v8h*)(hp + 16);
            a1[kq] = SHUF16(lo, hi);
        }

        // ---- layer 1: [16x64]x[64x32], same nt-outer scheme ----
#pragma unroll
        for (int nt = 0; nt < 2; ++nt) {
            v8f c[N_ELEM];
#pragma unroll
            for (int e = 0; e < N_ELEM; ++e) {
                v8f acc = {};
#pragma unroll
                for (int kq = 0; kq < 2; ++kq) {
                    const _Float16* bp = W1base + ((e * H1) + nt * 16 + mrow) * H0 + kq * 32 + koff;
                    v8h blo = *(const v8h*)bp;
                    v8h bhi = *(const v8h*)(bp + 16);
                    v16h b = SHUF16(blo, bhi);
                    acc = __builtin_amdgcn_wmma_f32_16x16x32_f16(
                              false, a1[kq], false, b, (short)0, acc, false, false);
                }
                c[e] = acc;
            }
            const int n = nt * 16 + mrow;
            const float bb0 = sB1[0 * H1 + n];
            const float bb1 = sB1[1 * H1 + n];
            const float bb2 = sB1[2 * H1 + n];
#pragma unroll
            for (int q = 0; q < 8; ++q) {
                const int e = er[q];
                float x = (e == 0 ? c[0][q] : (e == 1 ? c[1][q] : c[2][q]))
                        + (e == 0 ? bb0 : (e == 1 ? bb1 : bb2));
                sH2[wave][base8 + q][n] = softplus_fast(x);
            }
        }

        // ---- layer 2 (32 -> 1) + segment-mean accumulation ----
        if (lane < 16) {
            const int e = sElem[wave][mrow];
            float acc = sB2[e];
#pragma unroll
            for (int k = 0; k < H1; ++k)
                acc += sH2[wave][mrow][k] * sW2[e * H1 + k];
            const int m = molIdx[tileBase + mrow];
            atomicAdd(&sums[m], acc);
            atomicAdd(&counts[m], 1.0f);
        }
    }
}

// ---------------------------------------------------------------- launch

extern "C" void kernel_launch(void* const* d_in, const int* in_sizes, int n_in,
                              void* d_out, int out_size, void* d_ws, size_t ws_size,
                              hipStream_t stream) {
    const float* feats = (const float*)d_in[0];
    const int*   elem  = (const int*)d_in[1];
    const int*   mol   = (const int*)d_in[2];
    const float* W0    = (const float*)d_in[3];
    const float* b0    = (const float*)d_in[4];
    const float* W1    = (const float*)d_in[5];
    const float* b1    = (const float*)d_in[6];
    const float* W2    = (const float*)d_in[7];
    const float* b2    = (const float*)d_in[8];
    float* out = (float*)d_out;

    float*    sums   = (float*)d_ws;
    float*    counts = sums + N_MOLS;
    _Float16* W0t    = (_Float16*)((char*)d_ws + (size_t)2 * N_MOLS * sizeof(float));

    molnet_zero_ws<<<(2 * N_MOLS + 255) / 256, 256, 0, stream>>>(sums);
    molnet_prep_w0<<<(N_ELEM * D_IN * H0 + 255) / 256, 256, 0, stream>>>(W0, W0t);
    molnet_main<<<MAIN_BLOCKS, BLK_THREADS, 0, stream>>>(
        feats, elem, mol, W0t, W1, b0, b1, W2, b2, sums, counts);
    molnet_finalize<<<N_MOLS / 256, 256, 0, stream>>>(sums, counts, out);
}
